// ModelClass_48644799595233
// MI455X (gfx1250) — matrix operand order repaired and checked by reading, not compiled, not run on hardware
//
#include <hip/hip_runtime.h>

// ---------------- problem constants ----------------
#define BGR 64
#define NPLc 512
#define NLAY 3
#define NLn (BGR * NPLc)        // 32768 nodes per layer
#define NTOT (NLAY * NLn)       // 98304
#define E_INc (16 * NLn)        // 524288
#define E_FBc (8 * NLn)         // 262144
#define HID 256

// prepped-weight layout (elements, per MLP), kb-major within a layer:
//  frag(layer, kb, nt, lane) lives at ((kb*NT + nt)*32 + lane)*16
//  L0: Kp=32 ,Np=256 ->  8192   (KB=1, NT=16)
//  L1: Kp=256,Np=256 -> 65536   (KB=8, NT=16)  -> one kb-chunk = 8192 elems = 16KB
//  L2: Kp=256,Np=256 -> 65536
//  L3: Kp=256,Np=16  ->  4096   (KB=8, NT=1)
#define WL0 8192
#define WL1 65536
#define WL3 4096
#define WPER (WL0 + WL1 + WL1 + WL3)   // 143360 per MLP
#define BPER 784                        // 256+256+256+16 padded biases
#define NMLP 5                          // 0=inlayer 1=forward 2=backward 3=node 4=hlv
#define CHUNK 8192                      // bf16 elems per kb-chunk of a 256x256 layer

typedef __attribute__((ext_vector_type(16))) __bf16 v16bf;
typedef __attribute__((ext_vector_type(8)))  __bf16 v8bf;
typedef __attribute__((ext_vector_type(8)))  float  v8f;
typedef __attribute__((ext_vector_type(4)))  int    v4i;

#define ACT_STRIDE 264   // 256 + 8 bf16 pad -> row stride 528B, spreads LDS banks

// ---- CDNA5 async global->LDS copy (guarded; falls back to ld/ds_store) ----
#if defined(__has_builtin)
#if __has_builtin(__builtin_amdgcn_global_load_async_to_lds_b128)
#define HAVE_ASYNC_LDS 1
#endif
#endif

#ifdef HAVE_ASYNC_LDS
typedef v4i __attribute__((address_space(1)))* gv4p;   // global int4*
typedef v4i __attribute__((address_space(3)))* lv4p;   // LDS int4*
static __device__ inline void async_wait0() {
#if __has_builtin(__builtin_amdgcn_s_wait_asynccnt)
    __builtin_amdgcn_s_wait_asynccnt(0);
#else
    asm volatile("s_wait_asynccnt 0x0" ::: "memory");
#endif
}
#else
static __device__ inline void async_wait0() {}
#endif

// copy 64B per thread (256 threads -> one 16KB chunk)
static __device__ inline void stage_chunk(__bf16* dst, const __bf16* src, int tid) {
    const __bf16* s = src + tid * 32;
    __bf16* d = dst + tid * 32;
#ifdef HAVE_ASYNC_LDS
    gv4p gp = (gv4p)s;
    lv4p lp = (lv4p)d;
    __builtin_amdgcn_global_load_async_to_lds_b128(gp, lp, 0, 0);
    __builtin_amdgcn_global_load_async_to_lds_b128(gp, lp, 16, 0);
    __builtin_amdgcn_global_load_async_to_lds_b128(gp, lp, 32, 0);
    __builtin_amdgcn_global_load_async_to_lds_b128(gp, lp, 48, 0);
#else
    v8bf t0 = *(const v8bf*)(s);
    v8bf t1 = *(const v8bf*)(s + 8);
    v8bf t2 = *(const v8bf*)(s + 16);
    v8bf t3 = *(const v8bf*)(s + 24);
    *(v8bf*)(d)      = t0;
    *(v8bf*)(d + 8)  = t1;
    *(v8bf*)(d + 16) = t2;
    *(v8bf*)(d + 24) = t3;
#endif
}

// ---------------- weight/bias prep ----------------
struct PrepArgs {
    const float* W[20];
    const float* Bv[20];
    __bf16* wout;
    float*  bout;
};

__global__ __launch_bounds__(256) void prep_kernel(PrepArgs a) {
    long tid = (long)blockIdx.x * 256 + threadIdx.x;
    const long WTOT = (long)NMLP * WPER;
    if (tid < WTOT) {
        int m = (int)(tid / WPER);
        int r = (int)(tid % WPER);
        int layer, q;
        if (r < WL0)                { layer = 0; q = r;                 }
        else if (r < WL0 + WL1)     { layer = 1; q = r - WL0;           }
        else if (r < WL0 + 2 * WL1) { layer = 2; q = r - WL0 - WL1;     }
        else                        { layer = 3; q = r - WL0 - 2 * WL1; }
        int e    = q & 15;
        int lane = (q >> 4) & 31;
        int t    = q >> 9;
        int NT   = (layer == 3) ? 1 : 16;     // n-tiles per layer
        int kb   = t / NT;                    // kb-major chunks
        int nt   = t % NT;
        int K    = kb * 32 + ((lane >= 16) ? 8 : 0) + (e & 7) + ((e >> 3) << 4);   // B-frag K mapping
        int Ncol = nt * 16 + (lane & 15);                   // B-frag column = lane%16
        int din  = (layer == 0) ? ((m == 4) ? 10 : 14) : HID;
        int dout = (layer == 3) ? ((m == 4) ? 1 : 4) : HID;
        float v = 0.0f;
        if (K < din && Ncol < dout) v = a.W[m * 4 + layer][(long)K * dout + Ncol];
        a.wout[tid] = (__bf16)v;
    } else {
        long idx = tid - WTOT;
        if (idx < (long)NMLP * BPER) {
            int m = (int)(idx / BPER);
            int r = (int)(idx % BPER);
            int layer = (r < 256) ? 0 : (r < 512) ? 1 : (r < 768) ? 2 : 3;
            int n = r - layer * 256;
            int dout = (layer == 3) ? ((m == 4) ? 1 : 4) : HID;
            a.bout[idx] = (n < dout) ? a.Bv[m * 4 + layer][n] : 0.0f;
        }
    }
}

// ---------------- small utility kernels ----------------
__global__ void zerof_kernel(float* p, long n) {
    long i = (long)blockIdx.x * 256 + threadIdx.x;
    if (i < n) p[i] = 0.0f;
}

__global__ void init_x_kernel(const float* __restrict__ x0, float* __restrict__ X, int n) {
    int i = blockIdx.x * 256 + threadIdx.x;
    if (i < n) {
        X[i * 4 + 0] = x0[i];
        X[i * 4 + 1] = 0.0f;
        X[i * 4 + 2] = 0.0f;
        X[i * 4 + 3] = 0.0f;
    }
}

// F[i,0:4]=x  F[i,4:8]=static  F[i,8:14]=hlv[graph]  F[i,14:16]=0
__global__ void build_feat_kernel(const float* __restrict__ X, const float* __restrict__ stat,
                                  const float* __restrict__ hlvs, const int* __restrict__ gid,
                                  float* __restrict__ F, int base, int cnt) {
    int i = blockIdx.x * 256 + threadIdx.x;
    if (i >= cnt) return;
    int node = base + i;
    float* f = F + (long)i * 16;
    const float* xr = X + (long)node * 4;
    f[0] = xr[0]; f[1] = xr[1]; f[2] = xr[2]; f[3] = xr[3];
    const float* sr = stat + (long)node * 4;
    f[4] = sr[0]; f[5] = sr[1]; f[6] = sr[2]; f[7] = sr[3];
    const float* h = hlvs + (long)gid[node] * 6;
    f[8] = h[0]; f[9] = h[1]; f[10] = h[2]; f[11] = h[3]; f[12] = h[4]; f[13] = h[5];
    f[14] = 0.0f; f[15] = 0.0f;
}

// AGG[dst-dstOff] += F[src]  (14 real features)
__global__ void edge_agg_kernel(const float* __restrict__ F, const int* __restrict__ src,
                                const int* __restrict__ dst, int E, float* __restrict__ AGG,
                                int dstOff) {
    int e = blockIdx.x * 256 + threadIdx.x;
    if (e >= E) return;
    const float* s = F + (long)src[e] * 16;
    float* a = AGG + (long)(dst[e] - dstOff) * 16;
    #pragma unroll
    for (int c = 0; c < 14; ++c) atomicAdd(a + c, s[c]);
}

// XIN = (1+eps)*F + AGG  over NL*16 elements
__global__ void gin_input_kernel(const float* __restrict__ F, const float* __restrict__ AGG,
                                 const float* __restrict__ epsPtr, float* __restrict__ XIN, int n) {
    int i = blockIdx.x * 256 + threadIdx.x;
    if (i >= n) return;
    float e = 1.0f + epsPtr[0];
    XIN[i] = e * F[i] + AGG[i];
}

// HXIN[b,0:6]=hlvs  rest 0 (pool adds into 6:10 afterwards)
__global__ void hxin_init_kernel(const float* __restrict__ hlvs, float* __restrict__ HXIN) {
    int i = blockIdx.x * 256 + threadIdx.x;
    if (i >= BGR * 16) return;
    int b = i >> 4, c = i & 15;
    HXIN[i] = (c < 6) ? hlvs[b * 6 + c] : 0.0f;
}

__global__ void pool_kernel(const float* __restrict__ X, const int* __restrict__ gid,
                            float* __restrict__ HXIN, int n) {
    int i = blockIdx.x * 256 + threadIdx.x;
    if (i >= n) return;
    float* h = HXIN + (long)gid[i] * 16 + 6;
    #pragma unroll
    for (int c = 0; c < 4; ++c) atomicAdd(h + c, X[(long)i * 4 + c]);
}

// ---------------- fused 4-layer WMMA MLP ----------------
static __device__ inline v16bf ldfrag(const __bf16* p) {
    return *(const v16bf*)p;   // 32B contiguous fragment
}

// A-fragment (16x32 bf16) from LDS row-major activations
static __device__ inline v16bf ldact(const __bf16* row, int kb, int khalf) {
    v8bf lo = *(const v8bf*)(row + kb * 32 + khalf * 8);
    v8bf hi = *(const v8bf*)(row + kb * 32 + 16 + khalf * 8);
    v16bf r;
    #pragma unroll
    for (int i = 0; i < 8; ++i) { r[i] = lo[i]; r[8 + i] = hi[i]; }
    return r;
}

__global__ __launch_bounds__(256, 1) void mlp4_kernel(
    const float* __restrict__ xin,   // [M,16] f32 (K<=16 real, rest zero)
    float* __restrict__ out,         // row stride ostride, ocols real outputs
    const __bf16* __restrict__ wp,   // prepped bf16 weights (4 layers concat)
    const float* __restrict__ bp,    // prepped biases (256,256,256,16)
    int M, int ostride, int ocols)
{
    __shared__ __bf16 act[128 * ACT_STRIDE];     // 67584 B
    __shared__ __bf16 wbuf[2][CHUNK];            // 2 x 16 KB staged weights
    const int tid   = threadIdx.x;
    const int lane  = tid & 31;
    const int wave  = tid >> 5;
    const int m     = lane & 15;    // row within 16-row strip (A) / col within tile (C)
    const int khalf = lane >> 4;
    __bf16* arow = act + (long)(wave * 16) * ACT_STRIDE;
    const int rowbase = blockIdx.x * 128 + wave * 16;

    v8f acc[16];

    // ---- layer 0: K=32 (real K<=16), A built from f32 xin; weights direct from L2
    v16bf a0;
    {
        int row = rowbase + m;
        const float* src = xin + (long)row * 16;
        bool ok = (row < M);
        #pragma unroll
        for (int e = 0; e < 16; ++e) {
            int k = (e & 7) + ((e >> 3) << 4) + khalf * 8;   // A-frag K mapping
            float v = (ok && k < 16) ? src[k] : 0.0f;
            a0[e] = (__bf16)v;
        }
    }
    #pragma unroll
    for (int nt = 0; nt < 16; ++nt) {
        v16bf bf = ldfrag(wp + ((long)(nt * 32 + lane)) * 16);
        v8f z = {0.f,0.f,0.f,0.f,0.f,0.f,0.f,0.f};
        acc[nt] = __builtin_amdgcn_wmma_f32_16x16x32_bf16(false, a0, false, bf,
                                                          (short)0, z, false, false);
    }
    {
        const float* b0 = bp;
        #pragma unroll
        for (int nt = 0; nt < 16; ++nt) {
            float bv = b0[nt * 16 + m];
            #pragma unroll
            for (int j = 0; j < 8; ++j) {
                float v = fmaxf(acc[nt][j] + bv, 0.0f);
                arow[(khalf * 8 + j) * ACT_STRIDE + nt * 16 + m] = (__bf16)v;
            }
        }
    }
    __syncthreads();

    // ---- layers 1 and 2: 256x256, weights double-buffer staged in LDS (async)
    const __bf16* w = wp + WL0;
    const float*  b = bp + 256;
    #pragma unroll 1
    for (int layer = 0; layer < 2; ++layer) {
        #pragma unroll
        for (int nt = 0; nt < 16; ++nt) {
            v8f z = {0.f,0.f,0.f,0.f,0.f,0.f,0.f,0.f};
            acc[nt] = z;
        }
        stage_chunk(wbuf[0], w, tid);            // chunk 0 in flight
        int cur = 0;
        #pragma unroll 1
        for (int kb = 0; kb < 8; ++kb) {
            async_wait0();       // my async stores to LDS for chunk kb have landed
            __syncthreads();     // everyone's landed; previous buffer fully consumed
            if (kb < 7) stage_chunk(wbuf[cur ^ 1], w + (size_t)(kb + 1) * CHUNK, tid);
            v16bf a = ldact(arow + (long)m * ACT_STRIDE, kb, khalf);
            const __bf16* wl = wbuf[cur];
            #pragma unroll
            for (int nt = 0; nt < 16; ++nt) {
                v16bf bf = ldfrag(wl + (nt * 32 + lane) * 16);
                acc[nt] = __builtin_amdgcn_wmma_f32_16x16x32_bf16(false, a, false, bf,
                                                                  (short)0, acc[nt], false, false);
            }
            cur ^= 1;
        }
        __syncthreads();
        #pragma unroll
        for (int nt = 0; nt < 16; ++nt) {
            float bv = b[nt * 16 + m];
            #pragma unroll
            for (int j = 0; j < 8; ++j) {
                float v = fmaxf(acc[nt][j] + bv, 0.0f);
                arow[(khalf * 8 + j) * ACT_STRIDE + nt * 16 + m] = (__bf16)v;
            }
        }
        __syncthreads();
        w += WL1; b += 256;
    }

    // ---- layer 3: 256 -> 16 (ocols real); tiny weights direct from L2
    v8f accf = {0.f,0.f,0.f,0.f,0.f,0.f,0.f,0.f};
    #pragma unroll 1
    for (int kb = 0; kb < 8; ++kb) {
        v16bf a  = ldact(arow + (long)m * ACT_STRIDE, kb, khalf);
        v16bf bf = ldfrag(w + ((long)(kb * 32 + lane)) * 16);
        accf = __builtin_amdgcn_wmma_f32_16x16x32_bf16(false, a, false, bf,
                                                       (short)0, accf, false, false);
    }
    if (m < ocols) {
        float bv = b[m];
        #pragma unroll
        for (int j = 0; j < 8; ++j) {
            int row = rowbase + khalf * 8 + j;
            if (row < M) out[(long)row * ostride + m] = fmaxf(accf[j] + bv, 0.0f);
        }
    }
}

// ---------------- host driver ----------------
extern "C" void kernel_launch(void* const* d_in, const int* in_sizes, int n_in,
                              void* d_out, int out_size, void* d_ws, size_t ws_size,
                              hipStream_t stream)
{
    // ----- input index maps: handle both insertion-order and jax sorted-key flattening
    int ep_in, ep_fw, ep_bw;
    int wb[5]; // weight-base index for mlp m = {inlayer, forward, backward, node, hlv}
    if (in_sizes[18] == 1) {  // insertion order: inlayer,forward,backward,node,hlv
        ep_in = 0;  ep_fw = 9;  ep_bw = 18;
        wb[0] = 1;  wb[1] = 10; wb[2] = 19; wb[3] = 27; wb[4] = 35;
    } else {                  // sorted keys: backward,forward,hlv,inlayer,node
        ep_bw = 0;  ep_fw = 9;  ep_in = 26;
        wb[2] = 1;  wb[1] = 10; wb[4] = 18; wb[0] = 27; wb[3] = 35;
    }
    const float* eps_in = (const float*)d_in[ep_in];
    const float* eps_fw = (const float*)d_in[ep_fw];
    const float* eps_bw = (const float*)d_in[ep_bw];
    const float* x0   = (const float*)d_in[43];
    const float* stat = (const float*)d_in[44];
    const float* hlvs = (const float*)d_in[45];
    const int*   gid  = (const int*)d_in[46];
    const int* inner  = (const int*)d_in[47];
    const int* fedge  = (const int*)d_in[48];
    const int* bedge  = (const int*)d_in[49];

    // ----- workspace carve
    char* cur = (char*)d_ws;
    auto carve = [&](size_t bytes) { void* p = cur; cur += (bytes + 255) & ~(size_t)255; return p; };
    __bf16* WPREP = (__bf16*)carve((size_t)NMLP * WPER * sizeof(__bf16));
    float*  BPREP = (float*) carve((size_t)NMLP * BPER * sizeof(float));
    float*  X     = (float*) carve((size_t)NTOT * 4 * sizeof(float));
    float*  F     = (float*) carve((size_t)2 * NLn * 16 * sizeof(float));
    float*  AGG   = (float*) carve((size_t)NLn * 16 * sizeof(float));
    float*  XIN   = (float*) carve((size_t)NLn * 16 * sizeof(float));
    float*  HXIN  = (float*) carve((size_t)BGR * 16 * sizeof(float));
    (void)ws_size; (void)n_in; (void)out_size;

    // ----- weight/bias prep (every call; deterministic)
    PrepArgs pa;
    for (int m = 0; m < 5; ++m)
        for (int l = 0; l < 4; ++l) {
            pa.W[m * 4 + l]  = (const float*)d_in[wb[m] + 2 * l];
            pa.Bv[m * 4 + l] = (const float*)d_in[wb[m] + 2 * l + 1];
        }
    pa.wout = WPREP; pa.bout = BPREP;
    {
        long tot = (long)NMLP * WPER + (long)NMLP * BPER;
        prep_kernel<<<(int)((tot + 255) / 256), 256, 0, stream>>>(pa);
    }

    init_x_kernel<<<(NTOT + 255) / 256, 256, 0, stream>>>(x0, X, NTOT);

    auto zero = [&](float* p, long n) {
        zerof_kernel<<<(int)((n + 255) / 256), 256, 0, stream>>>(p, n);
    };
    auto buildF = [&](int base, int cnt) {
        build_feat_kernel<<<(cnt + 255) / 256, 256, 0, stream>>>(X, stat, hlvs, gid, F, base, cnt);
    };
    auto agg = [&](const int* s, const int* d, int E, int dstOff) {
        zero(AGG, (long)NLn * 16);
        edge_agg_kernel<<<(E + 255) / 256, 256, 0, stream>>>(F, s, d, E, AGG, dstOff);
    };
    auto ginXIN = [&](const float* Fofs, const float* epsp) {
        gin_input_kernel<<<(NLn * 16 + 255) / 256, 256, 0, stream>>>(Fofs, AGG, epsp, XIN, NLn * 16);
    };
    auto mlp = [&](const float* xin, float* out, int ostride, int ocols, int M, int midx) {
        mlp4_kernel<<<(M + 127) / 128, 256, 0, stream>>>(
            xin, out, WPREP + (size_t)midx * WPER, BPREP + (size_t)midx * BPER, M, ostride, ocols);
    };

    for (int p = 0; p < 2; ++p) {
        // forward sweep
        for (int l = 0; l < NLAY; ++l) {
            int lo = l * NLn;
            buildF(lo, NLn);
            agg(inner + (size_t)(2 * l) * E_INc, inner + (size_t)(2 * l + 1) * E_INc, E_INc, 0);
            ginXIN(F, eps_in);
            mlp(XIN, X + (size_t)lo * 4, 4, 4, NLn, 0);
            if (l < NLAY - 1) {
                buildF(lo, 2 * NLn);
                agg(fedge + (size_t)(2 * l) * E_FBc, fedge + (size_t)(2 * l + 1) * E_FBc, E_FBc, NLn);
                ginXIN(F + (size_t)NLn * 16, eps_fw);
                mlp(XIN, X + (size_t)(lo + NLn) * 4, 4, 4, NLn, 1);
                buildF(lo + NLn, NLn);
                mlp(F, X + (size_t)(lo + NLn) * 4, 4, 4, NLn, 3);
            }
        }
        // backward sweep
        for (int l = NLAY - 1; l >= 1; --l) {
            int lo = (l - 1) * NLn;
            buildF(lo, 2 * NLn);
            agg(bedge + (size_t)(2 * (l - 1)) * E_FBc, bedge + (size_t)(2 * (l - 1) + 1) * E_FBc,
                E_FBc, 0);
            ginXIN(F, eps_bw);
            mlp(XIN, X + (size_t)lo * 4, 4, 4, NLn, 2);
            buildF(lo, NLn);
            agg(inner + (size_t)(2 * (l - 1)) * E_INc, inner + (size_t)(2 * (l - 1) + 1) * E_INc,
                E_INc, 0);
            ginXIN(F, eps_in);
            mlp(XIN, X + (size_t)lo * 4, 4, 4, NLn, 0);
            buildF(lo, NLn);
            mlp(F, X + (size_t)lo * 4, 4, 4, NLn, 3);
        }
    }

    // global_add_pool + hlv MLP -> d_out [B,1] f32
    hxin_init_kernel<<<(BGR * 16 + 255) / 256, 256, 0, stream>>>(hlvs, HXIN);
    pool_kernel<<<(NTOT + 255) / 256, 256, 0, stream>>>(X, gid, HXIN, NTOT);
    mlp(HXIN, (float*)d_out, 1, 1, BGR, 4);
}
